// ResLSTMCell_17703855194306
// MI455X (gfx1250) — matrix operand order repaired and checked by reading, not compiled, not run on hardware
//
#include <hip/hip_runtime.h>
#include <hip/hip_bf16.h>
#include <math.h>

typedef __bf16 bf16;
typedef __attribute__((ext_vector_type(16))) __bf16 v16bf;
typedef __attribute__((ext_vector_type(8)))  float  v8f;
typedef __attribute__((ext_vector_type(4)))  int    v4i;

#define AS1 __attribute__((address_space(1)))
#define AS3 __attribute__((address_space(3)))

// ---------------- problem sizes ----------------
#define BATCH  16384
#define INPUTD 2048
#define HIDDEN 2048
#define PROJ   1024
#define K_XH   (INPUTD + PROJ)            // 3072
#define K_XHC  (INPUTD + PROJ + HIDDEN)   // 5120

// ---------------- GEMM tiling ----------------
#define BM 128
#define BN 128
#define BK 32
#define LDSS 40   // padded LDS row stride (bf16): 80B rows -> conflict-free b128 frag reads

// CDNA5 async memory->LDS path (GLOBAL_LOAD_ASYNC_TO_LDS_B128, ASYNCcnt)
#if defined(__has_builtin)
#  if __has_builtin(__builtin_amdgcn_global_load_async_to_lds_b128) && \
      __has_builtin(__builtin_amdgcn_s_wait_asynccnt)
#    define USE_ASYNC_LDS 1
#  endif
#endif
#ifndef USE_ASYNC_LDS
#  define USE_ASYNC_LDS 0
#endif

// =====================================================================
// bf16 GEMM:  C[M,N] (+)= A[M,K] * Bt[N,K]^T    (A row-major, Bt = W^T row-major)
// 256 threads (8 waves), tile 128x128, k-step 32, double-buffered LDS.
// Each wave: 64x32 sub-tile = 4x2 v_wmma_f32_16x16x32_bf16 accumulators.
// =====================================================================
__global__ __launch_bounds__(256) void gemm_bf16_wmma(
    const bf16* __restrict__ A,  int lda,
    const bf16* __restrict__ Bt, int ldb,
    float* __restrict__ C,       int ldc,
    int K, int accumulate)
{
    __shared__ __align__(16) bf16 As[2][BM * LDSS];
    __shared__ __align__(16) bf16 Bs[2][BN * LDSS];

    const int tid  = threadIdx.x;
    const int wid  = tid >> 5;
    const int lane = tid & 31;
    const int m0 = blockIdx.y * BM;
    const int n0 = blockIdx.x * BN;
    const int wm0 = (wid & 1) * 64;   // 2 waves along M
    const int wn0 = (wid >> 1) * 32;  // 4 waves along N

    // per-thread staging chunks: tile = 512 x 16B chunks, 2 per thread per matrix
    const int row0 = tid >> 2;              // 0..63
    const int col0 = (tid & 3) << 3;        // 0,8,16,24
    const int row1 = row0 + 64;             // 64..127
    const size_t ga0 = (size_t)(m0 + row0) * lda + col0;
    const size_t ga1 = (size_t)(m0 + row1) * lda + col0;
    const size_t gb0 = (size_t)(n0 + row0) * ldb + col0;
    const size_t gb1 = (size_t)(n0 + row1) * ldb + col0;
    const int la0 = row0 * LDSS + col0;
    const int la1 = row1 * LDSS + col0;

    v8f acc[4][2];
#pragma unroll
    for (int i = 0; i < 4; ++i)
#pragma unroll
        for (int j = 0; j < 2; ++j)
            acc[i][j] = (v8f){0.f,0.f,0.f,0.f,0.f,0.f,0.f,0.f};

    const int frag_half = lane >> 4;   // 0: lanes 0-15, 1: lanes 16-31
    const int frag_row  = lane & 15;

    int buf = 0;

#if USE_ASYNC_LDS
    // ---------------- async DMA staging (ASYNCcnt) ----------------
    #define STAGE_ASYNC(B_, K0_)                                                     \
        do {                                                                         \
            __builtin_amdgcn_global_load_async_to_lds_b128(                          \
                (AS1 v4i*)(A + ga0 + (K0_)), (AS3 v4i*)&As[B_][la0], 0, 0);          \
            __builtin_amdgcn_global_load_async_to_lds_b128(                          \
                (AS1 v4i*)(A + ga1 + (K0_)), (AS3 v4i*)&As[B_][la1], 0, 0);          \
            __builtin_amdgcn_global_load_async_to_lds_b128(                          \
                (AS1 v4i*)(Bt + gb0 + (K0_)), (AS3 v4i*)&Bs[B_][la0], 0, 0);         \
            __builtin_amdgcn_global_load_async_to_lds_b128(                          \
                (AS1 v4i*)(Bt + gb1 + (K0_)), (AS3 v4i*)&Bs[B_][la1], 0, 0);         \
        } while (0)

    STAGE_ASYNC(0, 0);
    for (int k0 = 0; k0 < K; k0 += BK) {
        __builtin_amdgcn_s_wait_asynccnt(0);   // my tile-t DMAs are in LDS
        __syncthreads();                       // everyone staged t; everyone done with t-1
        if (k0 + BK < K) STAGE_ASYNC(buf ^ 1, k0 + BK);   // overlap next DMA with compute
#else
    // ---------------- pipelined register staging fallback ----------------
    uint4 ra0, ra1, rb0, rb1;
    ra0 = *(const uint4*)(A  + ga0);
    ra1 = *(const uint4*)(A  + ga1);
    rb0 = *(const uint4*)(Bt + gb0);
    rb1 = *(const uint4*)(Bt + gb1);
    *(uint4*)(&As[0][la0]) = ra0;
    *(uint4*)(&As[0][la1]) = ra1;
    *(uint4*)(&Bs[0][la0]) = rb0;
    *(uint4*)(&Bs[0][la1]) = rb1;
    for (int k0 = 0; k0 < K; k0 += BK) {
        __syncthreads();                       // tile t staged; everyone done with t-1
        const bool more = (k0 + BK < K);
        if (more) {                            // issue next tile's loads; no wait here
            ra0 = *(const uint4*)(A  + ga0 + k0 + BK);
            ra1 = *(const uint4*)(A  + ga1 + k0 + BK);
            rb0 = *(const uint4*)(Bt + gb0 + k0 + BK);
            rb1 = *(const uint4*)(Bt + gb1 + k0 + BK);
        }
#endif

        // ---- fragments from LDS buffer `buf` ----
        // A 16x32 bf16: lanes0-15 hold K0..7 & K16..23; lanes16-31 hold K8..15 & K24..31
        v16bf af[4];
#pragma unroll
        for (int i = 0; i < 4; ++i) {
            const bf16* p = &As[buf][(wm0 + i * 16 + frag_row) * LDSS + frag_half * 8];
            ((uint4*)&af[i])[0] = *(const uint4*)(p);
            ((uint4*)&af[i])[1] = *(const uint4*)(p + 16);
        }
        // B 32x16 bf16: lanes0-15 (N=lane) hold K0..7 & K16..23; lanes16-31 hold K8..15 & K24..31
        v16bf bfr[2];
#pragma unroll
        for (int j = 0; j < 2; ++j) {
            const bf16* p = &Bs[buf][(wn0 + j * 16 + frag_row) * LDSS + frag_half * 8];
            ((uint4*)&bfr[j])[0] = *(const uint4*)(p);
            ((uint4*)&bfr[j])[1] = *(const uint4*)(p + 16);
        }

        // ---- 8 WMMAs ----
#pragma unroll
        for (int i = 0; i < 4; ++i)
#pragma unroll
            for (int j = 0; j < 2; ++j)
                acc[i][j] = __builtin_amdgcn_wmma_f32_16x16x32_bf16(
                    false, af[i], false, bfr[j], (short)0, acc[i][j], false, false);

#if !USE_ASYNC_LDS
        if (more) {                            // LDS stores after compute: loadcnt wait
            *(uint4*)(&As[buf ^ 1][la0]) = ra0;  // had a full WMMA phase to land
            *(uint4*)(&As[buf ^ 1][la1]) = ra1;
            *(uint4*)(&Bs[buf ^ 1][la0]) = rb0;
            *(uint4*)(&Bs[buf ^ 1][la1]) = rb1;
        }
#endif
        buf ^= 1;
    }

    // ---- C store: VGPR v -> M = v (+8 for lanes 16-31), N = lane&15 ----
#pragma unroll
    for (int i = 0; i < 4; ++i) {
#pragma unroll
        for (int j = 0; j < 2; ++j) {
            const int colg = n0 + wn0 + j * 16 + frag_row;
#pragma unroll
            for (int v = 0; v < 8; ++v) {
                const int rowg = m0 + wm0 + i * 16 + v + frag_half * 8;
                const size_t idx = (size_t)rowg * ldc + colg;
                float r = acc[i][j][v];
                if (accumulate) r += C[idx];
                C[idx] = r;
            }
        }
    }
}

// =====================================================================
// fp32 -> bf16 copy into a column slice of the packed activation matrix
// grid = (width/256, BATCH)
// =====================================================================
__global__ __launch_bounds__(256) void cvt_pack(
    const float* __restrict__ src, bf16* __restrict__ dst,
    int width, int dst_ld, int dst_off)
{
    const int j = blockIdx.x * 256 + threadIdx.x;
    const size_t b = blockIdx.y;
    dst[b * (size_t)dst_ld + dst_off + j] = (bf16)src[b * (size_t)width + j];
}

// =====================================================================
// fp32 [K,N] -> bf16 [N, dst_ld] transpose into row slice k_off..k_off+K
// block (32,8), grid (N/32, K/32); LDS 32x33 to kill bank conflicts
// =====================================================================
__global__ void transpose_cvt(
    const float* __restrict__ src, int K, int N,
    bf16* __restrict__ dst, int dst_ld, int k_off)
{
    __shared__ float tile[32][33];
    const int tx = threadIdx.x, ty = threadIdx.y;
    const int n = blockIdx.x * 32 + tx;
    const int kbase = blockIdx.y * 32;
#pragma unroll
    for (int i = 0; i < 4; ++i)
        tile[ty + i * 8][tx] = src[(size_t)(kbase + ty + i * 8) * N + n];
    __syncthreads();
#pragma unroll
    for (int i = 0; i < 4; ++i) {
        const int nr = blockIdx.x * 32 + ty + i * 8;
        const int kc = kbase + tx;
        dst[(size_t)nr * dst_ld + k_off + kc] = (bf16)tile[tx][ty + i * 8];
    }
}

// =====================================================================
// Epilogue 1: gates -> c_new (fp32 out) + bf16 c_new + bf16 tanh(c_new)
// out_c initially holds cellgate pre-activation; grid (HIDDEN/256, BATCH)
// =====================================================================
__global__ __launch_bounds__(256) void epilogue1(
    const float* __restrict__ ifg,     // [B, 2*HIDDEN]
    float* __restrict__ out_c,         // [B, HIDDEN] in/out
    const float* __restrict__ c_t,
    const float* __restrict__ bias_i,
    const float* __restrict__ bias_f,
    const float* __restrict__ bias_c,
    bf16* __restrict__ cnew_bf,
    bf16* __restrict__ rt_bf)
{
    const int n = blockIdx.x * 256 + threadIdx.x;
    const size_t b = blockIdx.y;
    const size_t idx = b * HIDDEN + n;
    float ig = ifg[b * (size_t)(2 * HIDDEN) + n] + bias_i[n];
    float fg = ifg[b * (size_t)(2 * HIDDEN) + HIDDEN + n] + bias_f[n];
    ig = 1.0f / (1.0f + __expf(-ig));
    fg = 1.0f / (1.0f + __expf(-fg));
    const float cg = tanhf(out_c[idx] + bias_c[n]);
    const float cn = fg * c_t[idx] + ig * cg;
    out_c[idx]   = cn;
    cnew_bf[idx] = (bf16)cn;
    rt_bf[idx]   = (bf16)tanhf(cn);
}

// =====================================================================
// Epilogue 2: h_new = sigmoid(opre + b_o) * racc ; grid (PROJ/256, BATCH)
// =====================================================================
__global__ __launch_bounds__(256) void epilogue2(
    float* __restrict__ out_h,          // holds opre, overwritten with h_new
    const float* __restrict__ racc,
    const float* __restrict__ bias_o)
{
    const int n = blockIdx.x * 256 + threadIdx.x;
    const size_t b = blockIdx.y;
    const size_t idx = b * PROJ + n;
    const float o = 1.0f / (1.0f + __expf(-(out_h[idx] + bias_o[n])));
    out_h[idx] = o * racc[idx];
}

// =====================================================================
extern "C" void kernel_launch(void* const* d_in, const int* in_sizes, int n_in,
                              void* d_out, int out_size, void* d_ws, size_t ws_size,
                              hipStream_t stream)
{
    (void)in_sizes; (void)n_in; (void)out_size; (void)ws_size;

    const float* x_t    = (const float*)d_in[0];
    const float* h_t    = (const float*)d_in[1];
    const float* c_t    = (const float*)d_in[2];
    const float* W_if_x = (const float*)d_in[3];
    const float* W_if_h = (const float*)d_in[4];
    const float* W_if_c = (const float*)d_in[5];
    const float* bias_i = (const float*)d_in[6];
    const float* bias_f = (const float*)d_in[7];
    const float* W_c_x  = (const float*)d_in[8];
    const float* W_c_h  = (const float*)d_in[9];
    const float* bias_c = (const float*)d_in[10];
    const float* W_o_x  = (const float*)d_in[11];
    const float* W_o_h  = (const float*)d_in[12];
    const float* W_o_c  = (const float*)d_in[13];
    const float* bias_o = (const float*)d_in[14];
    const float* W_r_p  = (const float*)d_in[15];
    const float* W_r_x  = (const float*)d_in[16];

    float* out_h = (float*)d_out;                       // [B, PROJ]
    float* out_c = out_h + (size_t)BATCH * PROJ;        // [B, HIDDEN]

    // ---------------- workspace carve-up ----------------
    char* p = (char*)d_ws;
    auto take = [&](size_t bytes) { char* q = p; p += (bytes + 255) & ~(size_t)255; return q; };
    bf16*  A_xhc = (bf16*)take((size_t)BATCH * K_XHC * 2);       // [B, x|h|c]
    bf16*  Wt_if = (bf16*)take((size_t)(2*HIDDEN) * K_XHC * 2);  // [4096, 5120]
    bf16*  Wt_c  = (bf16*)take((size_t)HIDDEN * K_XH * 2);       // [2048, 3072]
    bf16*  Wt_o  = (bf16*)take((size_t)PROJ   * K_XH * 2);       // [1024, 3072]
    bf16*  Wt_rx = (bf16*)take((size_t)PROJ * INPUTD * 2);       // [1024, 2048]
    bf16*  Wt_oc = (bf16*)take((size_t)PROJ * HIDDEN * 2);       // [1024, 2048]
    bf16*  Wt_rp = (bf16*)take((size_t)PROJ * HIDDEN * 2);       // [1024, 2048]
    float* ifg   = (float*)take((size_t)BATCH * 2 * HIDDEN * 4); // [B, 4096]
    float* racc  = (float*)take((size_t)BATCH * PROJ * 4);       // [B, 1024]
    bf16*  cn_bf = (bf16*)take((size_t)BATCH * HIDDEN * 2);
    bf16*  rt_bf = (bf16*)take((size_t)BATCH * HIDDEN * 2);

    // ---------------- pack activations: A = [x | h | c] bf16 ----------------
    cvt_pack<<<dim3(INPUTD/256, BATCH), 256, 0, stream>>>(x_t, A_xhc, INPUTD, K_XHC, 0);
    cvt_pack<<<dim3(PROJ/256,   BATCH), 256, 0, stream>>>(h_t, A_xhc, PROJ,   K_XHC, INPUTD);
    cvt_pack<<<dim3(HIDDEN/256, BATCH), 256, 0, stream>>>(c_t, A_xhc, HIDDEN, K_XHC, INPUTD + PROJ);

    // ---------------- transpose weights to bf16 [N, K] ----------------
    transpose_cvt<<<dim3((2*HIDDEN)/32, INPUTD/32), dim3(32,8), 0, stream>>>(W_if_x, INPUTD, 2*HIDDEN, Wt_if, K_XHC, 0);
    transpose_cvt<<<dim3((2*HIDDEN)/32, PROJ/32),   dim3(32,8), 0, stream>>>(W_if_h, PROJ,   2*HIDDEN, Wt_if, K_XHC, INPUTD);
    transpose_cvt<<<dim3((2*HIDDEN)/32, HIDDEN/32), dim3(32,8), 0, stream>>>(W_if_c, HIDDEN, 2*HIDDEN, Wt_if, K_XHC, INPUTD + PROJ);
    transpose_cvt<<<dim3(HIDDEN/32, INPUTD/32),     dim3(32,8), 0, stream>>>(W_c_x,  INPUTD, HIDDEN,   Wt_c,  K_XH, 0);
    transpose_cvt<<<dim3(HIDDEN/32, PROJ/32),       dim3(32,8), 0, stream>>>(W_c_h,  PROJ,   HIDDEN,   Wt_c,  K_XH, INPUTD);
    transpose_cvt<<<dim3(PROJ/32, INPUTD/32),       dim3(32,8), 0, stream>>>(W_o_x,  INPUTD, PROJ,     Wt_o,  K_XH, 0);
    transpose_cvt<<<dim3(PROJ/32, PROJ/32),         dim3(32,8), 0, stream>>>(W_o_h,  PROJ,   PROJ,     Wt_o,  K_XH, INPUTD);
    transpose_cvt<<<dim3(PROJ/32, INPUTD/32),       dim3(32,8), 0, stream>>>(W_r_x,  INPUTD, PROJ,     Wt_rx, INPUTD, 0);
    transpose_cvt<<<dim3(PROJ/32, HIDDEN/32),       dim3(32,8), 0, stream>>>(W_o_c,  HIDDEN, PROJ,     Wt_oc, HIDDEN, 0);
    transpose_cvt<<<dim3(PROJ/32, HIDDEN/32),       dim3(32,8), 0, stream>>>(W_r_p,  HIDDEN, PROJ,     Wt_rp, HIDDEN, 0);

    const dim3 blk(256);
    // ---------------- phase-A GEMMs ----------------
    // if_gates = [x|h|c] @ [W_if_x; W_if_h; W_if_c]   (K=5120, N=4096)
    gemm_bf16_wmma<<<dim3((2*HIDDEN)/BN, BATCH/BM), blk, 0, stream>>>(
        A_xhc, K_XHC, Wt_if, K_XHC, ifg, 2*HIDDEN, K_XHC, 0);
    // cellgate pre-act -> out_c region   (K=3072, N=2048)
    gemm_bf16_wmma<<<dim3(HIDDEN/BN, BATCH/BM), blk, 0, stream>>>(
        A_xhc, K_XHC, Wt_c, K_XH, out_c, HIDDEN, K_XH, 0);
    // outgate pre-act (x,h part) -> out_h region  (K=3072, N=1024)
    gemm_bf16_wmma<<<dim3(PROJ/BN, BATCH/BM), blk, 0, stream>>>(
        A_xhc, K_XHC, Wt_o, K_XH, out_h, PROJ, K_XH, 0);
    // residual: racc = x @ W_r_x          (K=2048, N=1024)
    gemm_bf16_wmma<<<dim3(PROJ/BN, BATCH/BM), blk, 0, stream>>>(
        A_xhc, K_XHC, Wt_rx, INPUTD, racc, PROJ, INPUTD, 0);

    // ---------------- gate nonlinearities, c_new ----------------
    epilogue1<<<dim3(HIDDEN/256, BATCH), blk, 0, stream>>>(
        ifg, out_c, c_t, bias_i, bias_f, bias_c, cn_bf, rt_bf);

    // ---------------- phase-B GEMMs (depend on c_new) ----------------
    // opre += c_new @ W_o_c
    gemm_bf16_wmma<<<dim3(PROJ/BN, BATCH/BM), blk, 0, stream>>>(
        cn_bf, HIDDEN, Wt_oc, HIDDEN, out_h, PROJ, HIDDEN, 1);
    // racc += tanh(c_new) @ W_r_proj
    gemm_bf16_wmma<<<dim3(PROJ/BN, BATCH/BM), blk, 0, stream>>>(
        rt_bf, HIDDEN, Wt_rp, HIDDEN, racc, PROJ, HIDDEN, 1);

    // ---------------- h_new ----------------
    epilogue2<<<dim3(PROJ/256, BATCH), blk, 0, stream>>>(out_h, racc, bias_o);
}
